// Attention_86131274154419
// MI455X (gfx1250) — matrix operand order repaired and checked by reading, not compiled, not run on hardware
//
#include <hip/hip_runtime.h>

#define HEADS 8
#define SCALE 0.17677669529663687f   // 32^-0.5

typedef __bf16 bf16_t;
typedef __attribute__((ext_vector_type(16))) __bf16 v16bf;
typedef __attribute__((ext_vector_type(8)))  float  v8f;

#define XSTR 264                         // xn row stride (bf16 elems), 256 + pad
#define QSTR 776                         // qkv row stride, 768 + pad
#define PSTR 40                          // P-stage row stride, 32 + pad

#define XP_BYTES   (HEADS * 64 * PSTR * 2)   // 40960 (also covers 64*XSTR*2 = 33792)
#define QKV_BYTES  (64 * QSTR * 2)           // 99328
#define BIAS_OFF   (XP_BYTES + QKV_BYTES)    // 140288
#define SMEM_BYTES (BIAS_OFF + 320 * 4)      // 141568

// d_ws layout (bytes)
#define WS_BIAS_OFF   0
#define WS_WQKV_OFF   4096                       // 48*8 frags * 1024 B = 393216
#define WS_WOUT_OFF   (WS_WQKV_OFF + 48 * 8 * 1024)   // 16*8 frags * 1024 B = 131072

// ---------------- WMMA helpers ----------------

__device__ __forceinline__ v8f wmma_bf16(v16bf a, v16bf b, v8f c) {
  return __builtin_amdgcn_wmma_f32_16x16x32_bf16(false, a, false, b, (short)0, c,
                                                 false, false);
}

// A-matrix 16x32 bf16 per-lane K index (contiguous, matches B): lanes0-15: v0..7 -> K0..15,
// lanes16-31: K16..31
__device__ __forceinline__ int a_k_idx(int v, int e, int lane) {
  int kb = (lane >> 4) << 4;
  return kb + 2 * v + e;
}

// Load A fragment (16 rows x 32 K) from row-major bf16 LDS: base -> [row0][k0]
__device__ __forceinline__ v16bf load_a_lds(const bf16_t* base, int stride, int lane) {
  const bf16_t* p = base + (lane & 15) * stride;
  v16bf a;
#pragma unroll
  for (int v = 0; v < 8; ++v) {
#pragma unroll
    for (int e = 0; e < 2; ++e) a[2 * v + e] = p[a_k_idx(v, e, lane)];
  }
  return a;
}

// B fragment from prepacked global weights: frag = ntile*KS + ks, 32 lanes x 16 bf16
// (32 contiguous bytes per lane -> two global_load_b128, fully coalesced)
__device__ __forceinline__ v16bf load_b_packed(const bf16_t* __restrict__ P, int frag,
                                               int lane) {
  const bf16_t* p = P + ((size_t)frag * 32 + lane) * 16;
  v16bf b;
#pragma unroll
  for (int i = 0; i < 16; ++i) b[i] = p[i];
  return b;
}

// B fragment where B[k][n] = lds[(n0+n)*stride + cbase + k]  (K^T access, for Q@K^T)
__device__ __forceinline__ v16bf load_b_ldsT(const bf16_t* lds, int stride, int n0,
                                             int cbase, int lane) {
  const bf16_t* p = lds + (n0 + (lane & 15)) * stride + cbase + ((lane >> 4) << 4);
  v16bf b;
#pragma unroll
  for (int i = 0; i < 16; ++i) b[i] = p[i];
  return b;
}

// B fragment where B[k][n] = lds[(k0+k)*stride + n0 + n]  (row-major, for P@V)
__device__ __forceinline__ v16bf load_b_lds(const bf16_t* lds, int stride, int k0,
                                            int n0, int lane) {
  int n = n0 + (lane & 15);
  int kg = k0 + ((lane >> 4) << 4);
  v16bf b;
#pragma unroll
  for (int v = 0; v < 8; ++v) {
    b[2 * v]     = lds[(kg + 2 * v) * stride + n];
    b[2 * v + 1] = lds[(kg + 2 * v + 1) * stride + n];
  }
  return b;
}

// ---------------- weight prepack: fp32 (K x N) -> bf16 B-fragment layout ----------------
// grid = NT*KS blocks, 32 threads; frag f = ntile*KS + ks

__global__ void pack_kernel(const float* __restrict__ W, int N, int KS,
                            bf16_t* __restrict__ P) {
  int f = blockIdx.x;
  int lane = threadIdx.x;
  int ntile = f / KS, ks = f % KS;
  int n = ntile * 16 + (lane & 15);
  int kg = ks * 32 + ((lane >> 4) << 4);
  bf16_t* dst = P + ((size_t)f * 32 + lane) * 16;
#pragma unroll
  for (int j = 0; j < 16; ++j) dst[j] = (bf16_t)W[(size_t)(kg + j) * N + n];
}

// ---------------- dynamic position bias MLP (tiny, one block) ----------------

__device__ void ln_relu64(float* h, const float* __restrict__ g,
                          const float* __restrict__ b) {
  float m = 0.f;
  for (int j = 0; j < 64; ++j) m += h[j];
  m *= (1.0f / 64.0f);
  float v = 0.f;
  for (int j = 0; j < 64; ++j) { float d = h[j] - m; v += d * d; }
  v *= (1.0f / 64.0f);
  float rs = rsqrtf(v + 1e-3f);
  for (int j = 0; j < 64; ++j) {
    float y = (h[j] - m) * rs * g[j] + b[j];
    h[j] = y > 0.f ? y : 0.f;
  }
}

__global__ void dpb_kernel(const float* __restrict__ w1, const float* __restrict__ b1,
                           const float* __restrict__ g1, const float* __restrict__ bb1,
                           const float* __restrict__ w2, const float* __restrict__ b2,
                           const float* __restrict__ g2, const float* __restrict__ bb2,
                           const float* __restrict__ w3, const float* __restrict__ b3,
                           const float* __restrict__ g3, const float* __restrict__ bb3,
                           const float* __restrict__ w4, const float* __restrict__ b4,
                           float* __restrict__ biases) {
  int t = threadIdx.x;
  if (t >= 289) return;
  float r0 = (float)(t / 17 - 8);
  float r1 = (float)(t % 17 - 8);
  float h[64], u[64];
  for (int j = 0; j < 64; ++j) h[j] = r0 * w1[j] + r1 * w1[64 + j] + b1[j];
  ln_relu64(h, g1, bb1);
  for (int j = 0; j < 64; ++j) {
    float s = b2[j];
    for (int k = 0; k < 64; ++k) s += h[k] * w2[k * 64 + j];
    u[j] = s;
  }
  ln_relu64(u, g2, bb2);
  for (int j = 0; j < 64; ++j) {
    float s = b3[j];
    for (int k = 0; k < 64; ++k) s += u[k] * w3[k * 64 + j];
    h[j] = s;
  }
  ln_relu64(h, g3, bb3);
  float o = b4[0];
  for (int k = 0; k < 64; ++k) o += h[k] * w4[k];
  biases[t] = o;
}

// ---------------- main fused windowed-attention kernel ----------------
// grid = 2048 windows (b=8, 16x16 windows); block = 256 threads = 8 waves

__global__ __launch_bounds__(256) void attn_kernel(
    const float* __restrict__ x, const float* __restrict__ ng,
    const float* __restrict__ nb, const bf16_t* __restrict__ pqkv,
    const bf16_t* __restrict__ pout, const float* __restrict__ bout,
    const float* __restrict__ biases, float* __restrict__ out) {
  extern __shared__ char smem[];
  bf16_t* xnp = (bf16_t*)smem;                    // xn (64x256), later P-stage (8 heads x 64x32)
  bf16_t* qkv = (bf16_t*)(smem + XP_BYTES);       // q|k|v (64x768), q cols reused for attn-out
  float*  biasLds = (float*)(smem + BIAS_OFF);

  const int tid = threadIdx.x;
  const int lane = tid & 31;
  const int wave = tid >> 5;
  const int blk = blockIdx.x;
  const int bb = blk >> 8;          // batch
  const int wh = (blk >> 4) & 15;   // window row
  const int ww = blk & 15;          // window col
  const int jlo = lane & 15;
  const int ihalf = (lane < 16) ? 0 : 8;

  for (int i = tid; i < 289; i += 256) biasLds[i] = biases[i];

  // ---- stage 0: LayerNorm (eps 1e-5) + convert to bf16 in LDS ----
  {
    int token = tid >> 2;          // 64 tokens, 4 lanes each
    int part = tid & 3;            // 64 channels per lane
    int s1 = token >> 3, s2 = token & 7;
    size_t goff = (((size_t)(bb * 128 + wh * 8 + s1)) * 128 + (ww * 8 + s2)) * 256 +
                  part * 64;
    const float4* xv = (const float4*)(x + goff);
    float4 vals[16];
    float s = 0.f, ss = 0.f;
#pragma unroll
    for (int j = 0; j < 16; ++j) {
      float4 t = xv[j];
      vals[j] = t;
      s += t.x + t.y + t.z + t.w;
      ss += t.x * t.x + t.y * t.y + t.z * t.z + t.w * t.w;
    }
    s += __shfl_xor(s, 1, 32);   s += __shfl_xor(s, 2, 32);
    ss += __shfl_xor(ss, 1, 32); ss += __shfl_xor(ss, 2, 32);
    float mean = s * (1.0f / 256.0f);
    float var = ss * (1.0f / 256.0f) - mean * mean;
    float rstd = rsqrtf(var + 1e-5f);
    const float4* gv = (const float4*)(ng + part * 64);
    const float4* bv = (const float4*)(nb + part * 64);
    bf16_t* dst = xnp + token * XSTR + part * 64;
#pragma unroll
    for (int j = 0; j < 16; ++j) {
      float4 t = vals[j], g4 = gv[j], c4 = bv[j];
      dst[4 * j + 0] = (bf16_t)((t.x - mean) * rstd * g4.x + c4.x);
      dst[4 * j + 1] = (bf16_t)((t.y - mean) * rstd * g4.y + c4.y);
      dst[4 * j + 2] = (bf16_t)((t.z - mean) * rstd * g4.z + c4.z);
      dst[4 * j + 3] = (bf16_t)((t.w - mean) * rstd * g4.w + c4.w);
    }
  }
  __syncthreads();

  // ---- stage 1: QKV GEMM (64x256)@(256x768) -> qkv LDS (q pre-scaled) ----
  {
    for (int nt = 0; nt < 6; ++nt) {
      int ntile = wave * 6 + nt;                   // 48 N-tiles total
      v8f acc[4];
#pragma unroll
      for (int mt = 0; mt < 4; ++mt)
#pragma unroll
        for (int r = 0; r < 8; ++r) acc[mt][r] = 0.f;
#pragma unroll
      for (int ks = 0; ks < 8; ++ks) {
        v16bf bf = load_b_packed(pqkv, ntile * 8 + ks, lane);
#pragma unroll
        for (int mt = 0; mt < 4; ++mt) {
          v16bf a = load_a_lds(xnp + mt * 16 * XSTR + ks * 32, XSTR, lane);
          acc[mt] = wmma_bf16(a, bf, acc[mt]);
        }
      }
      int col = ntile * 16 + jlo;
      float sc = (col < 256) ? SCALE : 1.0f;       // pre-scale q
#pragma unroll
      for (int mt = 0; mt < 4; ++mt) {
#pragma unroll
        for (int r = 0; r < 8; ++r)
          qkv[(mt * 16 + ihalf + r) * QSTR + col] = (bf16_t)(acc[mt][r] * sc);
      }
    }
  }
  __syncthreads();

  // ---- stage 2: per-head attention (wave = head) ----
  {
    const int h = wave;
    v16bf qa[4];
#pragma unroll
    for (int mt = 0; mt < 4; ++mt)
      qa[mt] = load_a_lds(qkv + mt * 16 * QSTR + h * 32, QSTR, lane);

    v8f sa[4][4];
#pragma unroll
    for (int mt = 0; mt < 4; ++mt)
#pragma unroll
      for (int nt = 0; nt < 4; ++nt)
#pragma unroll
        for (int r = 0; r < 8; ++r) sa[mt][nt][r] = 0.f;

#pragma unroll
    for (int nt = 0; nt < 4; ++nt) {
      v16bf kf = load_b_ldsT(qkv, QSTR, nt * 16, 256 + h * 32, lane);
#pragma unroll
      for (int mt = 0; mt < 4; ++mt) sa[mt][nt] = wmma_bf16(qa[mt], kf, sa[mt][nt]);
    }

    // bias add + row softmax (row i lives in one 16-lane half; reduce xor 1,2,4,8)
#pragma unroll
    for (int mt = 0; mt < 4; ++mt) {
#pragma unroll
      for (int r = 0; r < 8; ++r) {
        int i = mt * 16 + ihalf + r;
        int i1 = i >> 3, i2 = i & 7;
#pragma unroll
        for (int nt = 0; nt < 4; ++nt) {
          int j = nt * 16 + jlo;
          int d = (i1 - (j >> 3) + 7) * 15 + (i2 - (j & 7) + 7);
          sa[mt][nt][r] += biasLds[d];
        }
        float m = sa[mt][0][r];
#pragma unroll
        for (int nt = 1; nt < 4; ++nt) m = fmaxf(m, sa[mt][nt][r]);
        m = fmaxf(m, __shfl_xor(m, 1, 32));
        m = fmaxf(m, __shfl_xor(m, 2, 32));
        m = fmaxf(m, __shfl_xor(m, 4, 32));
        m = fmaxf(m, __shfl_xor(m, 8, 32));
        float s = 0.f;
#pragma unroll
        for (int nt = 0; nt < 4; ++nt) {
          float p = __expf(sa[mt][nt][r] - m);
          sa[mt][nt][r] = p;
          s += p;
        }
        s += __shfl_xor(s, 1, 32);
        s += __shfl_xor(s, 2, 32);
        s += __shfl_xor(s, 4, 32);
        s += __shfl_xor(s, 8, 32);
        float inv = 1.0f / s;
#pragma unroll
        for (int nt = 0; nt < 4; ++nt) sa[mt][nt][r] *= inv;
      }
    }

    // stage P (C-layout -> row-major LDS) for A-fragment reload
    bf16_t* pst = xnp + h * (64 * PSTR);
#pragma unroll
    for (int mt = 0; mt < 4; ++mt)
#pragma unroll
      for (int nt = 0; nt < 4; ++nt)
#pragma unroll
        for (int r = 0; r < 8; ++r)
          pst[(mt * 16 + ihalf + r) * PSTR + nt * 16 + jlo] = (bf16_t)sa[mt][nt][r];
    __syncthreads();

    // O = P (64x64) @ V (64x32)
    v8f o[4][2];
#pragma unroll
    for (int mt = 0; mt < 4; ++mt)
#pragma unroll
      for (int n2 = 0; n2 < 2; ++n2)
#pragma unroll
        for (int r = 0; r < 8; ++r) o[mt][n2][r] = 0.f;
#pragma unroll
    for (int kc = 0; kc < 2; ++kc) {
      v16bf pa[4], vf[2];
#pragma unroll
      for (int mt = 0; mt < 4; ++mt)
        pa[mt] = load_a_lds(pst + mt * 16 * PSTR + kc * 32, PSTR, lane);
#pragma unroll
      for (int n2 = 0; n2 < 2; ++n2)
        vf[n2] = load_b_lds(qkv, QSTR, kc * 32, 512 + h * 32 + n2 * 16, lane);
#pragma unroll
      for (int mt = 0; mt < 4; ++mt)
#pragma unroll
        for (int n2 = 0; n2 < 2; ++n2) o[mt][n2] = wmma_bf16(pa[mt], vf[n2], o[mt][n2]);
    }

    // store attn-out into q columns [h*32, h*32+32) of qkv (q no longer needed)
#pragma unroll
    for (int mt = 0; mt < 4; ++mt)
#pragma unroll
      for (int n2 = 0; n2 < 2; ++n2)
#pragma unroll
        for (int r = 0; r < 8; ++r)
          qkv[(mt * 16 + ihalf + r) * QSTR + h * 32 + n2 * 16 + jlo] =
              (bf16_t)o[mt][n2][r];
  }
  __syncthreads();

  // ---- stage 3: output projection (64x256)@(256x256) + bias, un-window store ----
  {
    for (int nt = 0; nt < 2; ++nt) {
      int ntile = wave * 2 + nt;                   // 16 N-tiles total
      v8f acc[4];
#pragma unroll
      for (int mt = 0; mt < 4; ++mt)
#pragma unroll
        for (int r = 0; r < 8; ++r) acc[mt][r] = 0.f;
#pragma unroll
      for (int ks = 0; ks < 8; ++ks) {
        v16bf bf = load_b_packed(pout, ntile * 8 + ks, lane);
#pragma unroll
        for (int mt = 0; mt < 4; ++mt) {
          v16bf a = load_a_lds(qkv + mt * 16 * QSTR + ks * 32, QSTR, lane);
          acc[mt] = wmma_bf16(a, bf, acc[mt]);
        }
      }
      int c = ntile * 16 + jlo;
      float bo = bout[c];
#pragma unroll
      for (int mt = 0; mt < 4; ++mt) {
#pragma unroll
        for (int r = 0; r < 8; ++r) {
          int i = mt * 16 + ihalf + r;
          int s1 = i >> 3, s2 = i & 7;
          size_t go =
              (((size_t)(bb * 128 + wh * 8 + s1)) * 128 + (ww * 8 + s2)) * 256 + c;
          out[go] = acc[mt][r] + bo;
        }
      }
    }
  }
}

// ---------------- launcher ----------------

extern "C" void kernel_launch(void* const* d_in, const int* in_sizes, int n_in,
                              void* d_out, int out_size, void* d_ws, size_t ws_size,
                              hipStream_t stream) {
  const float* x      = (const float*)d_in[0];
  const float* norm_g = (const float*)d_in[1];
  const float* norm_b = (const float*)d_in[2];
  const float* w_qkv  = (const float*)d_in[3];
  const float* w_out  = (const float*)d_in[4];
  const float* b_out  = (const float*)d_in[5];

  char* ws = (char*)d_ws;
  float*  biases = (float*)(ws + WS_BIAS_OFF);   // 289 floats
  bf16_t* pqkv   = (bf16_t*)(ws + WS_WQKV_OFF);  // 48*8 packed fragments
  bf16_t* pout   = (bf16_t*)(ws + WS_WOUT_OFF);  // 16*8 packed fragments

  dpb_kernel<<<1, 320, 0, stream>>>(
      (const float*)d_in[6], (const float*)d_in[7], (const float*)d_in[8],
      (const float*)d_in[9], (const float*)d_in[10], (const float*)d_in[11],
      (const float*)d_in[12], (const float*)d_in[13], (const float*)d_in[14],
      (const float*)d_in[15], (const float*)d_in[16], (const float*)d_in[17],
      (const float*)d_in[18], (const float*)d_in[19], biases);

  pack_kernel<<<48 * 8, 32, 0, stream>>>(w_qkv, 768, 8, pqkv);
  pack_kernel<<<16 * 8, 32, 0, stream>>>(w_out, 256, 8, pout);

  attn_kernel<<<2048, 256, SMEM_BYTES, stream>>>(x, norm_g, norm_b, pqkv, pout,
                                                 b_out, biases, (float*)d_out);
}
